// SelfAttention_17291538334039
// MI455X (gfx1250) — matrix operand order repaired and checked by reading, not compiled, not run on hardware
//
#include <hip/hip_runtime.h>
#include <hip/hip_bf16.h>

// Sigmoid self-attention for MI455X (gfx1250), wave32 + WMMA bf16 path.
//   q = x@Wq^T+bq ; k = x@Wk^T+bk ; v = x@Wv^T+bv
//   out = sigmoid((q k^T)/sqrt(D)) v
// All GEMMs on v_wmma_f32_16x16x32_bf16 (fp32 accumulate). All fp32->bf16
// conversion is hoisted into a one-shot elementwise pre-pass so the GEMM
// inner loops are pure {ds_load_b128 / global_load_b128 / v_wmma}.

#define BATCH 8
#define SEQ   2048
#define DIM   512

typedef __attribute__((ext_vector_type(16))) __bf16       v16bf;
typedef __attribute__((ext_vector_type(8)))  float        v8f;
typedef __attribute__((ext_vector_type(4)))  unsigned int u32x4;
typedef __attribute__((ext_vector_type(2)))  unsigned int u32x2;
typedef __attribute__((ext_vector_type(4)))  float        f32x4;

union Frag {
  v16bf          bf;
  unsigned short u[16];
  u32x4          q[2];
};

__device__ __forceinline__ unsigned short f2bf(float f) {
  unsigned int u = __builtin_bit_cast(unsigned int, f);
  u += 0x7FFFu + ((u >> 16) & 1u);          // round-to-nearest-even
  return (unsigned short)(u >> 16);
}

// LDS row strides: (stride_dwords mod 64) spreads 16 lanes over 16 banks,
// while remaining a multiple of 8 elements (16B) for u32x4 loads.
#define XPAD 520   // rows of 512 bf16, padded
#define VPAD 72    // rows of 64 bf16, padded

// ---------------------------------------------------------------------------
// Kernel 0: one-shot fp32 -> bf16 conversion of x and the three W matrices.
// ---------------------------------------------------------------------------
__global__ __launch_bounds__(256)
void cvt_kernel(const float* __restrict__ x,
                const float* __restrict__ wq, const float* __restrict__ wk,
                const float* __restrict__ wv,
                unsigned short* __restrict__ xbf,
                unsigned short* __restrict__ wbf)
{
  const size_t tid    = (size_t)blockIdx.x * 256 + threadIdx.x;
  const size_t stride = (size_t)gridDim.x * 256;

  // x: BATCH*SEQ*DIM elements, processed as f32x4 -> packed 2x u32
  const size_t NX4 = (size_t)BATCH * SEQ * DIM / 4;
  for (size_t i = tid; i < NX4; i += stride) {
    f32x4 f = ((const f32x4*)x)[i];
    u32x2 p;
    p[0] = (unsigned)f2bf(f[0]) | ((unsigned)f2bf(f[1]) << 16);
    p[1] = (unsigned)f2bf(f[2]) | ((unsigned)f2bf(f[3]) << 16);
    ((u32x2*)xbf)[i] = p;
  }

  // W: 3 matrices of DIM*DIM, packed contiguously into wbf
  const size_t NW4 = (size_t)DIM * DIM / 4;
  for (size_t i = tid; i < 3 * NW4; i += stride) {
    int m = (int)(i / NW4);
    size_t j = i - (size_t)m * NW4;
    const float* w = (m == 0) ? wq : (m == 1) ? wk : wv;
    f32x4 f = ((const f32x4*)w)[j];
    u32x2 p;
    p[0] = (unsigned)f2bf(f[0]) | ((unsigned)f2bf(f[1]) << 16);
    p[1] = (unsigned)f2bf(f[2]) | ((unsigned)f2bf(f[3]) << 16);
    ((u32x2*)wbf)[i] = p;
  }
}

// ---------------------------------------------------------------------------
// Kernel 1: fused QKV projection. blockIdx.y selects q/k/v.
//   A (16x32 bf16)  = x tile from LDS
//   B (32x16 bf16)  = W^T columns, straight bf16 global loads (32B/lane)
// Writes q,k as [B*SEQ][DIM] bf16; v transposed as [B][DIM][SEQ] bf16.
// ---------------------------------------------------------------------------
__global__ __launch_bounds__(256)
void qkv_proj_kernel(const unsigned short* __restrict__ xbf,
                     const unsigned short* __restrict__ wbf,
                     const float* __restrict__ bq, const float* __restrict__ bk,
                     const float* __restrict__ bv,
                     unsigned short* __restrict__ qbf,
                     unsigned short* __restrict__ kbf,
                     unsigned short* __restrict__ vtbf)
{
  __shared__ __align__(16) unsigned short xs[16 * XPAD];

  const int row0  = blockIdx.x * 16;   // flat token row over B*SEQ
  const int which = blockIdx.y;        // 0=q 1=k 2=v

  const unsigned short* w = wbf + (size_t)which * DIM * DIM;
  const float* bias = (which == 0) ? bq : (which == 1) ? bk : bv;

  // stage x tile (already bf16): 16 rows x 512, straight b128 copies
  {
    const u32x4* src = (const u32x4*)(xbf + (size_t)row0 * DIM);
    for (int i = threadIdx.x; i < 16 * (DIM / 8); i += 256) {
      int r = i >> 6, c = i & 63;            // 64 u32x4 per row
      ((u32x4*)(xs + r * XPAD))[c] = src[r * 64 + c];
    }
  }
  __syncthreads();

  const int wave = threadIdx.x >> 5;   // wave32
  const int lane = threadIdx.x & 31;
  const int hi   = lane >> 4;          // half-wave select for fragment layout
  const int lo   = lane & 15;

  for (int t = 0; t < 4; ++t) {
    const int e0 = (wave * 4 + t) * 16;          // output-column tile base
    const float bval = bias[e0 + lo];
    v8f acc;
#pragma unroll
    for (int r = 0; r < 8; ++r) acc[r] = bval;   // bias folded into C

    const unsigned short* wrow = w + (size_t)(e0 + lo) * DIM;
#pragma unroll 4
    for (int kk = 0; kk < DIM / 32; ++kk) {
      Frag a, b;
      const unsigned short* ap = xs + lo * XPAD + kk * 32 + hi * 16;
      a.q[0] = *(const u32x4*)(ap);
      a.q[1] = *(const u32x4*)(ap + 8);
      const u32x4* wp = (const u32x4*)(wrow + kk * 32 + hi * 16);
      b.q[0] = wp[0];
      b.q[1] = wp[1];
      acc = __builtin_amdgcn_wmma_f32_16x16x32_bf16(false, a.bf, false, b.bf,
                                                    (short)0, acc, false, false);
    }

    if (which == 2) {
      // store V transposed: [B][DIM][SEQ]
#pragma unroll
      for (int r = 0; r < 8; ++r) {
        int row = row0 + r + hi * 8;
        int bb = row >> 11, nn = row & (SEQ - 1);
        vtbf[((size_t)(bb * DIM + e0 + lo)) * SEQ + nn] = f2bf(acc[r]);
      }
    } else {
      unsigned short* o = (which == 0) ? qbf : kbf;
#pragma unroll
      for (int r = 0; r < 8; ++r) {
        int row = row0 + r + hi * 8;
        o[(size_t)row * DIM + e0 + lo] = f2bf(acc[r]);
      }
    }
  }
}

// ---------------------------------------------------------------------------
// Kernel 2: attention. One block = 16 queries x full DIM, 4 waves.
// Loop over 64-key blocks: each wave computes a 16x16 score tile (K=512),
// sigmoid in fp32, scores -> LDS bf16, then all waves accumulate S·V into
// 8 resident 16x16 fp32 accumulator tiles (128 output dims per wave).
// ---------------------------------------------------------------------------
__global__ __launch_bounds__(128)
void attn_kernel(const unsigned short* __restrict__ qbf,
                 const unsigned short* __restrict__ kbf,
                 const unsigned short* __restrict__ vtbf,
                 float* __restrict__ out)
{
  __shared__ __align__(16) unsigned short qs[16 * XPAD];   // ~16.6 KB
  __shared__ __align__(16) unsigned short vts[DIM * VPAD]; // ~73.7 KB  [d][key]
  __shared__ __align__(16) unsigned short ss[16 * VPAD];   // ~2.3 KB   [q][key]

  const int bb = blockIdx.x >> 7;            // batch
  const int q0 = (blockIdx.x & 127) * 16;    // query tile base

  const int wave = threadIdx.x >> 5;
  const int lane = threadIdx.x & 31;
  const int hi = lane >> 4, lo = lane & 15;

  // stage q tile (already bf16)
  {
    const u32x4* src = (const u32x4*)(qbf + ((size_t)(bb * SEQ + q0)) * DIM);
    for (int i = threadIdx.x; i < 16 * (DIM / 8); i += 128) {
      int r = i >> 6, c = i & 63;            // 64 u32x4 per row
      ((u32x4*)(qs + r * XPAD))[c] = src[r * 64 + c];
    }
  }

  v8f oacc[8];
#pragma unroll
  for (int t = 0; t < 8; ++t)
#pragma unroll
    for (int r = 0; r < 8; ++r) oacc[t][r] = 0.0f;

  const float scale = 0.044194173824159216f; // 1/sqrt(512)

  for (int kb = 0; kb < SEQ; kb += 64) {
    __syncthreads();  // previous iteration done reading vts/ss (and qs ready at kb=0)

    // stage transposed V tile: vts[d][0..63] <- vtbf[bb][d][kb..kb+63]
    // (contiguous 128B rows); prefetch next tile's row into L2 alongside.
    for (int i = threadIdx.x; i < DIM * 8; i += 128) {
      int d = i >> 3, c = i & 7;
      const unsigned short* src = vtbf + ((size_t)(bb * DIM + d)) * SEQ + kb + c * 8;
      ((u32x4*)(vts + d * VPAD))[c] = *(const u32x4*)src;
      if (kb + 64 < SEQ && c == 0)
        __builtin_prefetch(src + 64, 0, 3);  // global_prefetch_b8, next key block
    }

    // scores: wave handles keys [kb + wave*16, +16)
    v8f sacc;
#pragma unroll
    for (int r = 0; r < 8; ++r) sacc[r] = 0.0f;

    const unsigned short* krow =
        kbf + ((size_t)(bb * SEQ + kb + wave * 16 + lo)) * DIM;
    if (kb + 64 < SEQ)
      __builtin_prefetch(krow + (size_t)64 * DIM, 0, 3);  // next K rows
#pragma unroll 4
    for (int kk = 0; kk < DIM / 32; ++kk) {
      Frag a, b;
      const unsigned short* ap = qs + lo * XPAD + kk * 32 + hi * 16;
      a.q[0] = *(const u32x4*)(ap);
      a.q[1] = *(const u32x4*)(ap + 8);
      const u32x4* kp = (const u32x4*)(krow + kk * 32 + hi * 16);
      b.q[0] = kp[0];
      b.q[1] = kp[1];
      sacc = __builtin_amdgcn_wmma_f32_16x16x32_bf16(false, a.bf, false, b.bf,
                                                     (short)0, sacc, false, false);
    }

    // sigmoid in fp32, scores -> LDS as bf16
#pragma unroll
    for (int r = 0; r < 8; ++r) {
      float sv  = sacc[r] * scale;
      float sig = 1.0f / (1.0f + __expf(-sv));
      ss[(r + hi * 8) * VPAD + wave * 16 + lo] = f2bf(sig);
    }

    __syncthreads();  // scores + V tile visible to all waves

    // out += S(16x64) @ V(64x16) for this wave's 8 column tiles
#pragma unroll
    for (int t = 0; t < 8; ++t) {
      const int d0 = wave * 128 + t * 16;
#pragma unroll
      for (int step = 0; step < 2; ++step) {
        Frag a, b;
        const unsigned short* ap = ss + lo * VPAD + step * 32 + hi * 16;
        a.q[0] = *(const u32x4*)(ap);
        a.q[1] = *(const u32x4*)(ap + 8);
        const unsigned short* bp = vts + (d0 + lo) * VPAD + step * 32 + hi * 16;
        b.q[0] = ((const u32x4*)bp)[0];
        b.q[1] = ((const u32x4*)bp)[1];
        oacc[t] = __builtin_amdgcn_wmma_f32_16x16x32_bf16(false, a.bf, false, b.bf,
                                                          (short)0, oacc[t], false, false);
      }
    }
  }

  // epilogue: fp32 output [B][SEQ][DIM]
#pragma unroll
  for (int t = 0; t < 8; ++t) {
    const int d0 = wave * 128 + t * 16;
#pragma unroll
    for (int r = 0; r < 8; ++r) {
      size_t row = (size_t)bb * SEQ + q0 + r + hi * 8;
      out[row * DIM + d0 + lo] = oacc[t][r];
    }
  }
}

// ---------------------------------------------------------------------------
extern "C" void kernel_launch(void* const* d_in, const int* in_sizes, int n_in,
                              void* d_out, int out_size, void* d_ws, size_t ws_size,
                              hipStream_t stream) {
  (void)in_sizes; (void)n_in; (void)out_size; (void)ws_size;
  const float* x  = (const float*)d_in[0];
  const float* wq = (const float*)d_in[1];
  const float* wk = (const float*)d_in[2];
  const float* wv = (const float*)d_in[3];
  const float* bq = (const float*)d_in[4];
  const float* bk = (const float*)d_in[5];
  const float* bv = (const float*)d_in[6];
  float* out = (float*)d_out;

  // workspace: xbf (16MB) | wbf (1.5MB) | q (16MB) | k (16MB) | vT (16MB)
  unsigned short* xbf  = (unsigned short*)d_ws;
  unsigned short* wbf  = xbf + (size_t)BATCH * SEQ * DIM;
  unsigned short* qbf  = wbf + (size_t)3 * DIM * DIM;
  unsigned short* kbf  = qbf + (size_t)BATCH * SEQ * DIM;
  unsigned short* vtbf = kbf + (size_t)BATCH * SEQ * DIM;

  cvt_kernel<<<2048, 256, 0, stream>>>(x, wq, wk, wv, xbf, wbf);

  dim3 g1((BATCH * SEQ) / 16, 3);
  qkv_proj_kernel<<<g1, 256, 0, stream>>>(xbf, wbf, bq, bk, bv, qbf, kbf, vtbf);

  dim3 g2((BATCH * SEQ) / 16);
  attn_kernel<<<g2, 128, 0, stream>>>(qbf, kbf, vtbf, out);
}
